// VIBGATLayer_61426622267496
// MI455X (gfx1250) — matrix operand order repaired and checked by reading, not compiled, not run on hardware
//
#include <hip/hip_runtime.h>
#include <hip/hip_fp16.h>

#define Hd 128
#define Kn 32
#define Ld 64
#define LN_EPS 1e-5f
#define KL_BETA 0.001f

typedef __attribute__((ext_vector_type(16))) _Float16 v16h;
typedef __attribute__((ext_vector_type(8)))  _Float16 v8h;
typedef __attribute__((ext_vector_type(8)))  float    v8f;

// Assemble a 16-half WMMA fragment from two 16-byte chunks (LDS or global).
__device__ __forceinline__ v16h load_frag(const _Float16* lo, const _Float16* hi) {
    v8h a = *reinterpret_cast<const v8h*>(lo);
    v8h b = *reinterpret_cast<const v8h*>(hi);
    return __builtin_shufflevector(a, b, 0,1,2,3,4,5,6,7,8,9,10,11,12,13,14,15);
}

__device__ __forceinline__ v8f wmma_f16(v16h a, v16h b, v8f c) {
    return __builtin_amdgcn_wmma_f32_16x16x32_f16(false, a, false, b, (short)0, c, false, false);
}

// Async 16B global->LDS copy (CDNA5, ASYNCcnt-tracked). lds = wave-relative LDS
// byte offset (low 32 bits of a generic pointer to __shared__), g = global addr.
__device__ __forceinline__ void async_g2l_b128(unsigned lds, unsigned long long g) {
    asm volatile("global_load_async_to_lds_b128 %0, %1, off"
                 :: "v"(lds), "v"(g) : "memory");
}
__device__ __forceinline__ void wait_asynccnt0() {
    asm volatile("s_wait_asynccnt 0x0" ::: "memory");
}

// ---- prep: cast GEMM weights to f16 into workspace; zero the KL accumulator ----
__global__ void vib_prep(const float* Wmu, const float* Wlv, const float* Wk, const float* Wv,
                         _Float16* ws, float* klOut) {
    int i = blockIdx.x * blockDim.x + threadIdx.x;
    if (i == 0) *klOut = 0.0f;
    if (i < 8192)        ws[i] = (_Float16)Wmu[i];
    else if (i < 16384)  ws[i] = (_Float16)Wlv[i - 8192];
    else if (i < 20480)  ws[i] = (_Float16)Wk [i - 16384];
    else if (i < 24576)  ws[i] = (_Float16)Wv [i - 20480];
}

// ---- fused VIB-GAT: one workgroup (4 waves) per batch item ----
__global__ __launch_bounds__(128)
void vib_main(const float* __restrict__ self_feat, const float* __restrict__ neighbor,
              const float* __restrict__ trust,     const float* __restrict__ eps,
              const float* __restrict__ pre_g,     const float* __restrict__ pre_b,
              const float* __restrict__ b_mu,      const float* __restrict__ b_lv,
              const float* __restrict__ post_g,    const float* __restrict__ post_b,
              const float* __restrict__ W_q,       const float* __restrict__ b_q,
              const float* __restrict__ b_k,       const float* __restrict__ b_v,
              const float* __restrict__ W_o,       const float* __restrict__ b_o,
              const _Float16* __restrict__ Wmu16,  const _Float16* __restrict__ Wlv16,
              const _Float16* __restrict__ Wk16,   const _Float16* __restrict__ Wv16,
              float* __restrict__ out, float* __restrict__ klOut, int Btot)
{
    __shared__ _Float16 nnf16[Kn][Hd];   // normalized neighbors, f16 (8 KB)
    __shared__ float    zS[Kn][Ld];      // z staging f32, reused as k-matrix (8 KB)
    __shared__ _Float16 zf16[Kn][Ld];    // normalized z, f16 (4 KB)
    __shared__ float    vS[Kn][Ld];      // v-matrix (8 KB)
    __shared__ float    epsS[Kn][Ld];    // async-staged eps tile (8 KB)
    __shared__ float    qS[Ld];
    __shared__ float    ctxS[Ld];
    __shared__ float    attnS[Kn];
    __shared__ float2   redS[Kn][4];
    __shared__ float2   mrowS[Kn];
    __shared__ float    klred[128];

    const int b    = blockIdx.x;
    const int t    = threadIdx.x;
    const int lane = t & 31;
    const int wv   = t >> 5;          // nt: 16-column slice owned by this wave
    const int mloc = lane & 15;
    const int hi   = lane >> 4;
    const int nglob = wv * 16 + mloc; // output column of this lane's C tiles

    // ---- S-1: kick off async HBM->LDS copy of the 8KB contiguous eps tile.
    // Overlaps the pre-layernorm and the whole GEMM1 accumulation.
    {
        unsigned lbase = (unsigned)(size_t)(&epsS[0][0]) + (unsigned)(t * 16);
        unsigned long long gbase =
            (unsigned long long)(size_t)(eps + (size_t)b * Kn * Ld) + (unsigned)(t * 16);
        #pragma unroll
        for (int i = 0; i < 4; ++i)
            async_g2l_b128(lbase + i * 2048u, gbase + i * 2048ull);
    }

    // ---- S0: pre-layernorm of neighbor_feat [32,128] ----
    {
        const int r = t >> 2, qd = t & 3;
        const float* nrow = neighbor + ((size_t)b * Kn + r) * Hd + qd * 32;
        float sum = 0.f, ss = 0.f;
        #pragma unroll
        for (int c = 0; c < 32; c += 4) {
            float4 x = *reinterpret_cast<const float4*>(nrow + c);
            sum += x.x + x.y + x.z + x.w;
            ss  += x.x*x.x + x.y*x.y + x.z*x.z + x.w*x.w;
            nnf16[r][qd*32 + c + 0] = (_Float16)x.x;
            nnf16[r][qd*32 + c + 1] = (_Float16)x.y;
            nnf16[r][qd*32 + c + 2] = (_Float16)x.z;
            nnf16[r][qd*32 + c + 3] = (_Float16)x.w;
        }
        redS[r][qd] = make_float2(sum, ss);
        __syncthreads();
        if (t < Kn) {
            float s = 0.f, s2 = 0.f;
            #pragma unroll
            for (int q = 0; q < 4; ++q) { s += redS[t][q].x; s2 += redS[t][q].y; }
            float m = s / (float)Hd;
            float var = s2 / (float)Hd - m * m;
            mrowS[t] = make_float2(m, rsqrtf(var + LN_EPS));
        }
        __syncthreads();
        float m = mrowS[r].x, rs = mrowS[r].y;
        #pragma unroll 8
        for (int c = 0; c < 32; ++c) {
            int cc = qd*32 + c;
            float xv = (float)nnf16[r][cc];
            nnf16[r][cc] = (_Float16)((xv - m) * rs * pre_g[cc] + pre_b[cc]);
        }
        __syncthreads();
    }

    // ---- S1: q = self_feat @ W_q + b_q (1x128 @ 128x64) ----
    if (t < Ld) {
        const float* sf = self_feat + (size_t)b * Hd;
        float acc = b_q[t];
        for (int h = 0; h < Hd; ++h) acc += sf[h] * W_q[h * Ld + t];
        qS[t] = acc;
    }

    // ---- S2: GEMM1 mu/logvar = nn @ W_mu / W_lv (32x128 @ 128x64), WMMA f16 ----
    float klacc = 0.f;
    {
        v8f z8 = {0.f,0.f,0.f,0.f,0.f,0.f,0.f,0.f};
        v8f accMu[2] = { z8, z8 };
        v8f accLv[2] = { z8, z8 };
        #pragma unroll
        for (int kk = 0; kk < 4; ++kk) {
            const int kb = kk * 32 + hi * 8;
            v16h a0 = load_frag(&nnf16[mloc     ][kb], &nnf16[mloc     ][kb + 16]);
            v16h a1 = load_frag(&nnf16[mloc + 16][kb], &nnf16[mloc + 16][kb + 16]);
            const int krow = kk * 32 + hi * 8;
            const _Float16* wm = Wmu16 + (size_t)krow * Ld + wv * 16 + mloc;
            const _Float16* wl = Wlv16 + (size_t)krow * Ld + wv * 16 + mloc;
            v16h bm, bl;
            #pragma unroll
            for (int s = 0; s < 16; ++s) {
                const int ko = (s < 8) ? s : s + 8;
                bm[s] = wm[(size_t)ko * Ld];
                bl[s] = wl[(size_t)ko * Ld];
            }
            accMu[0] = wmma_f16(a0, bm, accMu[0]);
            accMu[1] = wmma_f16(a1, bm, accMu[1]);
            accLv[0] = wmma_f16(a0, bl, accLv[0]);
            accLv[1] = wmma_f16(a1, bl, accLv[1]);
        }
        // eps tile must be fully resident & visible to all threads before use
        wait_asynccnt0();
        __syncthreads();
        // epilogue: reparameterize z = mu + eps*exp(0.5*lv), accumulate KL terms
        const float bm = b_mu[nglob], bl = b_lv[nglob];
        #pragma unroll
        for (int mt = 0; mt < 2; ++mt) {
            #pragma unroll
            for (int v = 0; v < 8; ++v) {
                int m = mt * 16 + v + 8 * hi;
                float mu = accMu[mt][v] + bm;
                float lv = accLv[mt][v] + bl;
                float e  = epsS[m][nglob];
                klacc += 1.0f + lv - mu * mu - __expf(lv);
                zS[m][nglob] = mu + e * __expf(0.5f * lv);
            }
        }
        __syncthreads();
    }

    // ---- S3: post-layernorm of z [32,64] -> zf16 ----
    {
        const int r = t >> 2, qd = t & 3;
        float sum = 0.f, ss = 0.f;
        #pragma unroll
        for (int c = 0; c < 16; ++c) {
            float v = zS[r][qd*16 + c];
            sum += v; ss += v * v;
        }
        redS[r][qd] = make_float2(sum, ss);
        __syncthreads();
        if (t < Kn) {
            float s = 0.f, s2 = 0.f;
            #pragma unroll
            for (int q = 0; q < 4; ++q) { s += redS[t][q].x; s2 += redS[t][q].y; }
            float m = s / (float)Ld;
            float var = s2 / (float)Ld - m * m;
            mrowS[t] = make_float2(m, rsqrtf(var + LN_EPS));
        }
        __syncthreads();
        float m = mrowS[r].x, rs = mrowS[r].y;
        #pragma unroll
        for (int c = 0; c < 16; ++c) {
            int cc = qd*16 + c;
            zf16[r][cc] = (_Float16)((zS[r][cc] - m) * rs * post_g[cc] + post_b[cc]);
        }
        __syncthreads();
    }

    // ---- S4: GEMM2 k/v = z @ W_k / W_v (32x64 @ 64x64), WMMA f16 ----
    {
        v8f z8 = {0.f,0.f,0.f,0.f,0.f,0.f,0.f,0.f};
        v8f accK[2] = { z8, z8 };
        v8f accV[2] = { z8, z8 };
        #pragma unroll
        for (int kk = 0; kk < 2; ++kk) {
            const int kb = kk * 32 + hi * 8;
            v16h a0 = load_frag(&zf16[mloc     ][kb], &zf16[mloc     ][kb + 16]);
            v16h a1 = load_frag(&zf16[mloc + 16][kb], &zf16[mloc + 16][kb + 16]);
            const int krow = kk * 32 + hi * 8;
            const _Float16* wk  = Wk16 + (size_t)krow * Ld + wv * 16 + mloc;
            const _Float16* wvp = Wv16 + (size_t)krow * Ld + wv * 16 + mloc;
            v16h bk, bv;
            #pragma unroll
            for (int s = 0; s < 16; ++s) {
                const int ko = (s < 8) ? s : s + 8;
                bk[s] = wk[(size_t)ko * Ld];
                bv[s] = wvp[(size_t)ko * Ld];
            }
            accK[0] = wmma_f16(a0, bk, accK[0]);
            accK[1] = wmma_f16(a1, bk, accK[1]);
            accV[0] = wmma_f16(a0, bv, accV[0]);
            accV[1] = wmma_f16(a1, bv, accV[1]);
        }
        const float bk = b_k[nglob], bv = b_v[nglob];
        #pragma unroll
        for (int mt = 0; mt < 2; ++mt) {
            #pragma unroll
            for (int v = 0; v < 8; ++v) {
                int m = mt * 16 + v + 8 * hi;
                zS[m][nglob] = accK[mt][v] + bk;   // zS reused as k-matrix
                vS[m][nglob] = accV[mt][v] + bv;
            }
        }
        __syncthreads();
    }

    // ---- S5: trust-masked logits + softmax over K=32 (wave 0) ----
    if (t < Kn) {
        float dot = 0.f;
        #pragma unroll 16
        for (int c = 0; c < Ld; ++c) dot += qS[c] * zS[t][c];
        float lg = dot * 0.125f + __logf(trust[(size_t)b * Kn + t] + 1e-6f);
        float mx = lg;
        #pragma unroll
        for (int off = 16; off > 0; off >>= 1) mx = fmaxf(mx, __shfl_xor(mx, off, 32));
        float e = __expf(lg - mx);
        float s = e;
        #pragma unroll
        for (int off = 16; off > 0; off >>= 1) s += __shfl_xor(s, off, 32);
        attnS[t] = e / s;
    }
    __syncthreads();

    // ---- S6: context = attn . v ; comm = context @ W_o + b_o ----
    if (t < Ld) {
        float acc = 0.f;
        #pragma unroll 8
        for (int m = 0; m < Kn; ++m) acc += attnS[m] * vS[m][t];
        ctxS[t] = acc;
    }
    __syncthreads();
    {
        float acc = b_o[t];
        #pragma unroll 8
        for (int n = 0; n < Ld; ++n) acc += ctxS[n] * W_o[n * Hd + t];
        out[(size_t)b * Hd + t] = acc;
    }

    // ---- S7: KL reduction ----
    klred[t] = klacc;
    __syncthreads();
    if (t == 0) {
        float s = 0.f;
        #pragma unroll 8
        for (int i = 0; i < 128; ++i) s += klred[i];
        atomicAdd(klOut, s * (-0.5f * KL_BETA / (float)((size_t)Btot * Kn)));
    }
}

extern "C" void kernel_launch(void* const* d_in, const int* in_sizes, int n_in,
                              void* d_out, int out_size, void* d_ws, size_t ws_size,
                              hipStream_t stream) {
    const float* self_feat = (const float*)d_in[0];
    const float* neighbor  = (const float*)d_in[1];
    const float* trust     = (const float*)d_in[2];
    const float* eps       = (const float*)d_in[3];
    const float* pre_g     = (const float*)d_in[4];
    const float* pre_b     = (const float*)d_in[5];
    const float* W_mu      = (const float*)d_in[6];
    const float* b_mu      = (const float*)d_in[7];
    const float* W_lv      = (const float*)d_in[8];
    const float* b_lv      = (const float*)d_in[9];
    const float* post_g    = (const float*)d_in[10];
    const float* post_b    = (const float*)d_in[11];
    const float* W_q       = (const float*)d_in[12];
    const float* b_q       = (const float*)d_in[13];
    const float* W_k       = (const float*)d_in[14];
    const float* b_k       = (const float*)d_in[15];
    const float* W_v       = (const float*)d_in[16];
    const float* b_v       = (const float*)d_in[17];
    const float* W_o       = (const float*)d_in[18];
    const float* b_o       = (const float*)d_in[19];

    const int Btot = in_sizes[0] / Hd;
    float* out   = (float*)d_out;
    float* klOut = out + (size_t)Btot * Hd;

    _Float16* ws16 = (_Float16*)d_ws;
    const _Float16* Wmu16 = ws16;
    const _Float16* Wlv16 = ws16 + 8192;
    const _Float16* Wk16  = ws16 + 16384;
    const _Float16* Wv16  = ws16 + 20480;

    vib_prep<<<96, 256, 0, stream>>>(W_mu, W_lv, W_k, W_v, ws16, klOut);
    vib_main<<<Btot, 128, 0, stream>>>(self_feat, neighbor, trust, eps,
                                       pre_g, pre_b, b_mu, b_lv, post_g, post_b,
                                       W_q, b_q, b_k, b_v, W_o, b_o,
                                       Wmu16, Wlv16, Wk16, Wv16,
                                       out, klOut, Btot);
}
